// MoE_14293651161236
// MI455X (gfx1250) — compile-verified, matrix operand order checked by reading, not run on hardware
//
#include <hip/hip_runtime.h>
#include <hip/hip_bf16.h>

#define NTOK   1024
#define D_DIM  1024
#define H_DIM  1024
#define NEXP   8
#define TM     128
#define TN     128
#define TK     32
#define LDT    40            // bf16 elements per LDS row (pad: 16B-aligned rows)
#define NTHREADS 512         // 16 waves
#define MAX_RSLOTS 3072      // 2048 routed slots + per-expert pad to 128
#define SHARED_BASE_ROW MAX_RSLOTS

typedef __attribute__((ext_vector_type(16))) __bf16 v16bf;
typedef __attribute__((ext_vector_type(2)))  __bf16 v2bf;
typedef __attribute__((ext_vector_type(8)))  float  v8f;

// 16-byte int vector type matching the async-LDS builtin's pointer params.
typedef int gv4i __attribute__((vector_size(16)));
typedef __attribute__((address_space(1))) gv4i* as1_v4i_ptr;
typedef __attribute__((address_space(3))) gv4i* as3_v4i_ptr;

#if __has_builtin(__builtin_amdgcn_global_load_async_to_lds_b128)
#define HAS_ASYNC_LDS 1
#else
#define HAS_ASYNC_LDS 0
#endif

__device__ __forceinline__ void wait_async0() {
#if __has_builtin(__builtin_amdgcn_s_wait_asynccnt)
  __builtin_amdgcn_s_wait_asynccnt(0);
#else
  asm volatile("s_wait_asynccnt 0x0" ::: "memory");
#endif
}

__device__ __forceinline__ unsigned pack_bf16(float a, float b) {
#if __has_builtin(__builtin_amdgcn_cvt_pk_bf16_f32)
  v2bf v = __builtin_amdgcn_cvt_pk_bf16_f32(a, b);
  return __builtin_bit_cast(unsigned, v);
#else
  unsigned short ua = __builtin_bit_cast(unsigned short, (__bf16)a);
  unsigned short ub = __builtin_bit_cast(unsigned short, (__bf16)b);
  return (unsigned)ua | ((unsigned)ub << 16);
#endif
}

// Build a 16x32 bf16 WMMA fragment for one lane from an LDS row (K-contiguous).
// lanes 0-15 (hi=0): VGPR0..3 = K{0..7}, VGPR4..7 = K{16..23};
// lanes 16-31 (hi=1): K{8..15} and K{24..31}.  -> two b128 LDS loads.
__device__ __forceinline__ v16bf load_frag(const __bf16* row, int hi) {
  const unsigned* p = (const unsigned*)row;
  union { unsigned u[8]; v16bf v; } f;
  const int b = hi * 4;
  f.u[0] = p[b + 0]; f.u[1] = p[b + 1]; f.u[2] = p[b + 2]; f.u[3] = p[b + 3];
  f.u[4] = p[8 + b + 0]; f.u[5] = p[8 + b + 1]; f.u[6] = p[8 + b + 2]; f.u[7] = p[8 + b + 3];
  return f.v;
}

// Load 2 K-rows x 4 N-cols of an fp32 weight, pack to bf16 K-pairs (regs).
__device__ __forceinline__ void wload4(const float* __restrict__ W, int ld,
                                       int krow, int ncol, unsigned o[4], bool pf) {
  const float* pa = W + (size_t)krow * ld + ncol;
  if (pf) __builtin_prefetch(pa + (size_t)2 * TK * ld, 0, 3);
  float4 r0 = *(const float4*)pa;
  float4 r1 = *(const float4*)(pa + ld);
  o[0] = pack_bf16(r0.x, r1.x); o[1] = pack_bf16(r0.y, r1.y);
  o[2] = pack_bf16(r0.z, r1.z); o[3] = pack_bf16(r0.w, r1.w);
}
__device__ __forceinline__ void wstore4(__bf16* dst, const unsigned o[4]) {
#pragma unroll
  for (int j = 0; j < 4; j++) *(unsigned*)&dst[j * LDT] = o[j];
}

// ---------------- Router: logits -> softmax -> top-2 (+bias for select) -----
__global__ __launch_bounds__(256) void moe_router(
    const float* __restrict__ x, const float* __restrict__ rw,
    const float* __restrict__ ebias, int* __restrict__ sel,
    float* __restrict__ topw) {
  const int wave = threadIdx.x >> 5;
  const int lane = threadIdx.x & 31;
  const int t = blockIdx.x * 8 + wave;
  float acc[NEXP];
#pragma unroll
  for (int e = 0; e < NEXP; e++) acc[e] = 0.f;
  const float* xr = x + (size_t)t * D_DIM;
  for (int i = lane; i < D_DIM; i += 32) {
    const float xv = xr[i];
    const float* r = rw + (size_t)i * NEXP;
#pragma unroll
    for (int e = 0; e < NEXP; e++) acc[e] += xv * r[e];
  }
#pragma unroll
  for (int e = 0; e < NEXP; e++)
    for (int o = 16; o > 0; o >>= 1) acc[e] += __shfl_xor(acc[e], o, 32);
  if (lane == 0) {
    float m = acc[0];
#pragma unroll
    for (int e = 1; e < NEXP; e++) m = fmaxf(m, acc[e]);
    float p[NEXP], s = 0.f;
#pragma unroll
    for (int e = 0; e < NEXP; e++) { p[e] = __expf(acc[e] - m); s += p[e]; }
    const float inv = 1.f / s;
#pragma unroll
    for (int e = 0; e < NEXP; e++) p[e] *= inv;
    int b0 = 0, b1 = -1;
    float v0 = -3.4e38f, v1 = -3.4e38f;
#pragma unroll
    for (int e = 0; e < NEXP; e++) {
      const float q = p[e] + ebias[e];
      if (q > v0) { v1 = v0; b1 = b0; v0 = q; b0 = e; }
      else if (q > v1) { v1 = q; b1 = e; }
    }
    sel[t * 2 + 0] = b0; sel[t * 2 + 1] = b1;
    topw[t * 2 + 0] = p[b0]; topw[t * 2 + 1] = p[b1];
  }
}

// ---------------- Slot assignment (counting sort, 128-row padded segments) --
__global__ __launch_bounds__(1024) void moe_assign(
    const int* __restrict__ sel, int* __restrict__ cnt, int* __restrict__ off,
    int* __restrict__ tok2slot, int* __restrict__ slot_token) {
  __shared__ int scnt[NEXP];
  __shared__ int soff[NEXP];
  const int tid = threadIdx.x;
  if (tid < NEXP) scnt[tid] = 0;
  for (int i = tid; i < MAX_RSLOTS; i += 1024) slot_token[i] = -1;
  __syncthreads();
  const int e0 = sel[tid * 2 + 0];
  const int e1 = sel[tid * 2 + 1];
  const int p0 = atomicAdd(&scnt[e0], 1);
  const int p1 = atomicAdd(&scnt[e1], 1);
  __syncthreads();
  if (tid == 0) {
    int run = 0;
    for (int e = 0; e < NEXP; e++) {
      soff[e] = run;
      run += ((scnt[e] + TM - 1) / TM) * TM;
    }
  }
  __syncthreads();
  const int s0 = soff[e0] + p0, s1 = soff[e1] + p1;
  slot_token[s0] = tid; slot_token[s1] = tid;
  tok2slot[tid * 2 + 0] = s0; tok2slot[tid * 2 + 1] = s1;
  if (tid < NEXP) { cnt[tid] = scnt[tid]; off[tid] = soff[tid]; }
}

// ---------------- GEMM1: h = silu(x@w1) * (x@w3), double-buffered bf16 WMMA -
__global__ __launch_bounds__(NTHREADS) void moe_gemm1(
    const float* __restrict__ x, const float* __restrict__ w1,
    const float* __restrict__ w3, const float* __restrict__ sw1,
    const float* __restrict__ sw3, const int* __restrict__ cnt,
    const int* __restrict__ off, const int* __restrict__ slot_token,
    __bf16* __restrict__ h) {
  const int e = blockIdx.z, mt = blockIdx.y, nb = blockIdx.x;
  int rows, base;
  const float *W1, *W3;
  if (e < NEXP) {
    rows = cnt[e]; base = off[e];
    W1 = w1 + (size_t)e * D_DIM * H_DIM;
    W3 = w3 + (size_t)e * D_DIM * H_DIM;
  } else {
    rows = NTOK; base = SHARED_BASE_ROW; W1 = sw1; W3 = sw3;
  }
  if (mt * TM >= rows) return;

  __shared__ __bf16 sA[2][TM * LDT];
  __shared__ __bf16 sB1[2][TN * LDT];
  __shared__ __bf16 sB3[2][TN * LDT];
  __shared__ int tokRow[TM];

  const int tid = threadIdx.x;
  if (tid < TM) {
    const int r = mt * TM + tid;
    tokRow[tid] = (e < NEXP) ? slot_token[base + r] : r;
  }
  __syncthreads();

  const int lane = tid & 31, wv = tid >> 5;
  const int wm = wv & 3, wn = wv >> 2;             // 4x4 wave grid
  const int hi = lane >> 4, ln = lane & 15;
  const int arow = tid >> 2, akc = (tid & 3) * 8;  // A staging: 128 rows x 32k
  const int k0 = (tid >> 5) * 2, n0 = (tid & 31) * 4;  // B staging map
  const int ncol0 = nb * TN;
  const int myTok = tokRow[arow];

  const v8f vz = {0.f, 0.f, 0.f, 0.f, 0.f, 0.f, 0.f, 0.f};
  v8f acc1[2][2], acc3[2][2];
#pragma unroll
  for (int mf = 0; mf < 2; mf++)
#pragma unroll
    for (int nf = 0; nf < 2; nf++) { acc1[mf][nf] = vz; acc3[mf][nf] = vz; }

  // ---- A tile loader (gathered token rows, fp32 -> packed bf16 regs) ----
  auto a_load = [&](int kk) -> uint4 {
    float va[8];
    if (myTok >= 0) {
      const float* p = x + (size_t)myTok * D_DIM + kk + akc;
      float4 f0 = *(const float4*)p;
      float4 f1 = *(const float4*)(p + 4);
      va[0] = f0.x; va[1] = f0.y; va[2] = f0.z; va[3] = f0.w;
      va[4] = f1.x; va[5] = f1.y; va[6] = f1.z; va[7] = f1.w;
    } else {
#pragma unroll
      for (int j = 0; j < 8; j++) va[j] = 0.f;
    }
    uint4 pk;
    pk.x = pack_bf16(va[0], va[1]); pk.y = pack_bf16(va[2], va[3]);
    pk.z = pack_bf16(va[4], va[5]); pk.w = pack_bf16(va[6], va[7]);
    return pk;
  };

  const int NK = D_DIM / TK;
  // Prologue: stage 0
  {
    uint4 apk = a_load(0);
    unsigned b1r[4], b3r[4];
    wload4(W1, H_DIM, k0, ncol0 + n0, b1r, true);
    wload4(W3, H_DIM, k0, ncol0 + n0, b3r, true);
    *(uint4*)&sA[0][arow * LDT + akc] = apk;
    wstore4(&sB1[0][n0 * LDT + k0], b1r);
    wstore4(&sB3[0][n0 * LDT + k0], b3r);
  }
  __syncthreads();

  int s = 0;
  for (int kt = 0; kt < NK; kt++) {
    const int nk = kt + 1;
    uint4 apk;
    unsigned b1r[4], b3r[4];
    if (nk < NK) {
      const int kk = nk * TK;
      apk = a_load(kk);
      const bool pf = (kk + 2 * TK) < D_DIM;
      wload4(W1, H_DIM, kk + k0, ncol0 + n0, b1r, pf);
      wload4(W3, H_DIM, kk + k0, ncol0 + n0, b3r, pf);
    }
    // compute on stage s
    {
      const __bf16* bufA = sA[s];
      const v16bf af0 = load_frag(&bufA[(wm * 32 + ln) * LDT], hi);
      const v16bf af1 = load_frag(&bufA[(wm * 32 + 16 + ln) * LDT], hi);
#pragma unroll
      for (int nf = 0; nf < 2; nf++) {
        const v16bf b1f = load_frag(&sB1[s][(wn * 32 + nf * 16 + ln) * LDT], hi);
        acc1[0][nf] = __builtin_amdgcn_wmma_f32_16x16x32_bf16(
            false, af0, false, b1f, (short)0, acc1[0][nf], false, false);
        acc1[1][nf] = __builtin_amdgcn_wmma_f32_16x16x32_bf16(
            false, af1, false, b1f, (short)0, acc1[1][nf], false, false);
        const v16bf b3f = load_frag(&sB3[s][(wn * 32 + nf * 16 + ln) * LDT], hi);
        acc3[0][nf] = __builtin_amdgcn_wmma_f32_16x16x32_bf16(
            false, af0, false, b3f, (short)0, acc3[0][nf], false, false);
        acc3[1][nf] = __builtin_amdgcn_wmma_f32_16x16x32_bf16(
            false, af1, false, b3f, (short)0, acc3[1][nf], false, false);
      }
    }
    if (nk < NK) {
      *(uint4*)&sA[s ^ 1][arow * LDT + akc] = apk;
      wstore4(&sB1[s ^ 1][n0 * LDT + k0], b1r);
      wstore4(&sB3[s ^ 1][n0 * LDT + k0], b3r);
    }
    __syncthreads();
    s ^= 1;
  }

  // Epilogue: silu(g)*u -> bf16 h
#pragma unroll
  for (int mf = 0; mf < 2; mf++) {
    const int rowl_base = wm * 32 + mf * 16 + 8 * hi;
#pragma unroll
    for (int nf = 0; nf < 2; nf++) {
      const int col = ncol0 + wn * 32 + nf * 16 + ln;
#pragma unroll
      for (int r = 0; r < 8; r++) {
        const float g = acc1[mf][nf][r];
        const float u = acc3[mf][nf][r];
        const float sig = 1.0f / (1.0f + __expf(-g));
        const float hv = g * sig * u;
        const size_t grow = (size_t)(base + mt * TM + rowl_base + r);
        h[grow * H_DIM + col] = (__bf16)hv;
      }
    }
  }
}

// ---------------- GEMM2: y = h @ w2 (bf16 A via async LDS copy), fp32 out ---
__global__ __launch_bounds__(NTHREADS) void moe_gemm2(
    const __bf16* __restrict__ h, const float* __restrict__ w2,
    const float* __restrict__ sw2, const int* __restrict__ cnt,
    const int* __restrict__ off, float* __restrict__ y_routed,
    float* __restrict__ shared_y) {
  const int e = blockIdx.z, mt = blockIdx.y, nb = blockIdx.x;
  int rows, base;
  const float* W2;
  float* ybase;
  if (e < NEXP) {
    rows = cnt[e]; base = off[e];
    W2 = w2 + (size_t)e * H_DIM * D_DIM;
    ybase = y_routed + (size_t)base * D_DIM;
  } else {
    rows = NTOK; base = SHARED_BASE_ROW; W2 = sw2; ybase = shared_y;
  }
  if (mt * TM >= rows) return;

  __shared__ __bf16 sA[2][TM * LDT];
  __shared__ __bf16 sBT[2][TN * LDT];

  const int tid = threadIdx.x;
  const int lane = tid & 31, wv = tid >> 5;
  const int wm = wv & 3, wn = wv >> 2;
  const int hi = lane >> 4, ln = lane & 15;
  const int arow = tid >> 2, akc = (tid & 3) * 8;
  const int k0 = (tid >> 5) * 2, n0 = (tid & 31) * 4;
  const int ncol0 = nb * TN;
  const int rowbase = base + mt * TM;
  const __bf16* aRowPtr = h + (size_t)(rowbase + arow) * H_DIM + akc;

  const v8f vz = {0.f, 0.f, 0.f, 0.f, 0.f, 0.f, 0.f, 0.f};
  v8f acc[2][2];
#pragma unroll
  for (int mf = 0; mf < 2; mf++)
#pragma unroll
    for (int nf = 0; nf < 2; nf++) acc[mf][nf] = vz;

  const int NK = H_DIM / TK;
  // Prologue: stage 0
  {
#if HAS_ASYNC_LDS
    __builtin_amdgcn_global_load_async_to_lds_b128(
        (as1_v4i_ptr)(aRowPtr),
        (as3_v4i_ptr)&sA[0][arow * LDT + akc], 0, 0);
#else
    *(uint4*)&sA[0][arow * LDT + akc] = *(const uint4*)aRowPtr;
#endif
    unsigned br[4];
    wload4(W2, D_DIM, k0, ncol0 + n0, br, true);
    wstore4(&sBT[0][n0 * LDT + k0], br);
#if HAS_ASYNC_LDS
    wait_async0();
#endif
  }
  __syncthreads();

  int s = 0;
  for (int kt = 0; kt < NK; kt++) {
    const int nk = kt + 1;
    unsigned br[4];
#if !HAS_ASYNC_LDS
    uint4 apk;
#endif
    if (nk < NK) {
      const int kk = nk * TK;
#if HAS_ASYNC_LDS
      __builtin_amdgcn_global_load_async_to_lds_b128(
          (as1_v4i_ptr)(aRowPtr + kk),
          (as3_v4i_ptr)&sA[s ^ 1][arow * LDT + akc], 0, 0);
#else
      apk = *(const uint4*)(aRowPtr + kk);
#endif
      const bool pf = (kk + 2 * TK) < H_DIM;
      wload4(W2, D_DIM, kk + k0, ncol0 + n0, br, pf);
    }
    // compute on stage s
    {
      const __bf16* bufA = sA[s];
      const v16bf af0 = load_frag(&bufA[(wm * 32 + ln) * LDT], hi);
      const v16bf af1 = load_frag(&bufA[(wm * 32 + 16 + ln) * LDT], hi);
#pragma unroll
      for (int nf = 0; nf < 2; nf++) {
        const v16bf bfr = load_frag(&sBT[s][(wn * 32 + nf * 16 + ln) * LDT], hi);
        acc[0][nf] = __builtin_amdgcn_wmma_f32_16x16x32_bf16(
            false, af0, false, bfr, (short)0, acc[0][nf], false, false);
        acc[1][nf] = __builtin_amdgcn_wmma_f32_16x16x32_bf16(
            false, af1, false, bfr, (short)0, acc[1][nf], false, false);
      }
    }
    if (nk < NK) {
#if !HAS_ASYNC_LDS
      *(uint4*)&sA[s ^ 1][arow * LDT + akc] = apk;
#endif
      wstore4(&sBT[s ^ 1][n0 * LDT + k0], br);
    }
#if HAS_ASYNC_LDS
    wait_async0();
#endif
    __syncthreads();
    s ^= 1;
  }

#pragma unroll
  for (int mf = 0; mf < 2; mf++) {
    const int rowl_base = wm * 32 + mf * 16 + 8 * hi;
#pragma unroll
    for (int nf = 0; nf < 2; nf++) {
      const int col = ncol0 + wn * 32 + nf * 16 + ln;
#pragma unroll
      for (int r = 0; r < 8; r++)
        ybase[(size_t)(mt * TM + rowl_base + r) * D_DIM + col] = acc[mf][nf][r];
    }
  }
}

// ---------------- Combine: out = shared + w0*y[s0] + w1*y[s1] --------------
__global__ __launch_bounds__(256) void moe_combine(
    const float* __restrict__ shared_y, const float* __restrict__ y_routed,
    const int* __restrict__ tok2slot, const float* __restrict__ topw,
    float* __restrict__ out) {
  const int t = blockIdx.x;
  const int d = threadIdx.x * 4;
  const int s0 = tok2slot[t * 2 + 0], s1 = tok2slot[t * 2 + 1];
  const float w0 = topw[t * 2 + 0], w1 = topw[t * 2 + 1];
  float4 sh = *(const float4*)(shared_y + (size_t)t * D_DIM + d);
  float4 a  = *(const float4*)(y_routed + (size_t)s0 * D_DIM + d);
  float4 b  = *(const float4*)(y_routed + (size_t)s1 * D_DIM + d);
  float4 o;
  o.x = sh.x + w0 * a.x + w1 * b.x;
  o.y = sh.y + w0 * a.y + w1 * b.y;
  o.z = sh.z + w0 * a.z + w1 * b.z;
  o.w = sh.w + w0 * a.w + w1 * b.w;
  *(float4*)(out + (size_t)t * D_DIM + d) = o;
}

extern "C" void kernel_launch(void* const* d_in, const int* in_sizes, int n_in,
                              void* d_out, int out_size, void* d_ws,
                              size_t ws_size, hipStream_t stream) {
  (void)in_sizes; (void)n_in; (void)out_size; (void)ws_size;
  const float* x     = (const float*)d_in[0];
  const float* rw    = (const float*)d_in[1];
  const float* ebias = (const float*)d_in[2];
  const float* w1    = (const float*)d_in[3];
  const float* w2    = (const float*)d_in[4];
  const float* w3    = (const float*)d_in[5];
  const float* sw1   = (const float*)d_in[6];
  const float* sw2   = (const float*)d_in[7];
  const float* sw3   = (const float*)d_in[8];
  float* out = (float*)d_out;

  char* ws = (char*)d_ws;
  size_t cur = 0;
  auto take = [&](size_t bytes) {
    size_t p = cur;
    cur += (bytes + 255) & ~(size_t)255;
    return p;
  };
  int*    cnt        = (int*)(ws + take(NEXP * sizeof(int)));
  int*    offp       = (int*)(ws + take(NEXP * sizeof(int)));
  int*    sel        = (int*)(ws + take((size_t)NTOK * 2 * sizeof(int)));
  float*  topw       = (float*)(ws + take((size_t)NTOK * 2 * sizeof(float)));
  int*    tok2slot   = (int*)(ws + take((size_t)NTOK * 2 * sizeof(int)));
  int*    slot_token = (int*)(ws + take((size_t)MAX_RSLOTS * sizeof(int)));
  __bf16* hbuf       = (__bf16*)(ws + take((size_t)(MAX_RSLOTS + NTOK) * H_DIM * sizeof(__bf16)));
  float*  y_routed   = (float*)(ws + take((size_t)MAX_RSLOTS * D_DIM * sizeof(float)));
  float*  shared_y   = (float*)(ws + take((size_t)NTOK * D_DIM * sizeof(float)));

  moe_router<<<NTOK / 8, 256, 0, stream>>>(x, rw, ebias, sel, topw);
  moe_assign<<<1, 1024, 0, stream>>>(sel, cnt, offp, tok2slot, slot_token);
  moe_gemm1<<<dim3(H_DIM / TN, 16, NEXP + 1), NTHREADS, 0, stream>>>(
      x, w1, w3, sw1, sw3, cnt, offp, slot_token, hbuf);
  moe_gemm2<<<dim3(D_DIM / TN, 16, NEXP + 1), NTHREADS, 0, stream>>>(
      hbuf, w2, sw2, cnt, offp, y_routed, shared_y);
  moe_combine<<<NTOK, 256, 0, stream>>>(shared_y, y_routed, tok2slot, topw, out);
}
